// Attention_49821620634264
// MI455X (gfx1250) — compile-verified, matrix operand order checked by reading, not run on hardware
//
#include <hip/hip_runtime.h>
#include <hip/hip_bf16.h>

#define BB 2
#define LL 2048
#define DD 1024
#define HH 16
#define DEPTH 64
#define NEG_BIG (-1000000000.0f)

typedef __attribute__((ext_vector_type(16))) __bf16 v16bf;
typedef __attribute__((ext_vector_type(8)))  float  v8f;

#define WMMA_BF16(a, b, c) \
  __builtin_amdgcn_wmma_f32_16x16x32_bf16(false, (a), false, (b), (short)0, (c), false, false)

static __device__ __forceinline__ __bf16 f2bf(float f) { return (__bf16)f; }

// Load 8 contiguous bf16 (16B) into fragment slots [dst..dst+7]
static __device__ __forceinline__ void ld8bf(v16bf& f, int dst, const __bf16* p) {
  union { uint4 u; __bf16 h[8]; } t;
  t.u = *(const uint4*)p;
#pragma unroll
  for (int i = 0; i < 8; ++i) f[dst + i] = t.h[i];
}

// Load 8 contiguous fp32, convert to bf16 into fragment slots [dst..dst+7]
static __device__ __forceinline__ void ld8f(v16bf& f, int dst, const float* p) {
  float4 a = *(const float4*)p;
  float4 b = *(const float4*)(p + 4);
  f[dst + 0] = f2bf(a.x); f[dst + 1] = f2bf(a.y);
  f[dst + 2] = f2bf(a.z); f[dst + 3] = f2bf(a.w);
  f[dst + 4] = f2bf(b.x); f[dst + 5] = f2bf(b.y);
  f[dst + 6] = f2bf(b.z); f[dst + 7] = f2bf(b.w);
}

// reductions across the 16 lanes of a wave32 half (xor masks < 16 stay in-half)
static __device__ __forceinline__ float redmax16(float v) {
#pragma unroll
  for (int m = 1; m < 16; m <<= 1) v = fmaxf(v, __shfl_xor(v, m, 32));
  return v;
}
static __device__ __forceinline__ float redsum16(float v) {
#pragma unroll
  for (int m = 1; m < 16; m <<= 1) v += __shfl_xor(v, m, 32);
  return v;
}

// ---------------------------------------------------------------------------
// Kernel 1: transpose + convert the four DxD fp32 weights into bf16 W^T
// (channel-major: WT[n][k] = W[k][n]) so WMMA B-fragments load contiguously.
// ---------------------------------------------------------------------------
__global__ void __launch_bounds__(256) transpose_w(
    const float* __restrict__ W0, const float* __restrict__ W1,
    const float* __restrict__ W2, const float* __restrict__ W3,
    __bf16* __restrict__ out) {
  const float* W = (blockIdx.z == 0) ? W0 : (blockIdx.z == 1) ? W1
                 : (blockIdx.z == 2) ? W2 : W3;
  __bf16* O = out + (size_t)blockIdx.z * DD * DD;
  __shared__ float tile[32][33];
  int tx = threadIdx.x, ty = threadIdx.y;
  int x0 = blockIdx.x * 32;  // column (n) range
  int y0 = blockIdx.y * 32;  // row (k) range
#pragma unroll
  for (int i = ty; i < 32; i += 8)
    tile[i][tx] = W[(size_t)(y0 + i) * DD + x0 + tx];
  __syncthreads();
#pragma unroll
  for (int c = ty; c < 32; c += 8)
    O[(size_t)(x0 + c) * DD + y0 + tx] = f2bf(tile[tx][c]);
}

// ---------------------------------------------------------------------------
// Kernel 2: projection GEMM  OUT = bf16( X[4096x1024] @ W )  via W^T bf16.
// Each wave computes a 16(M) x 64(N) tile, K-loop step 32 (wmma 16x16x32).
// vt_mode=0: store [B,H,L,depth] (Q,K).  vt_mode=1: store [B,H,depth,L] (V^T).
// ---------------------------------------------------------------------------
__global__ void __launch_bounds__(256) qkv_gemm(
    const float* __restrict__ X, const __bf16* __restrict__ WT,
    __bf16* __restrict__ OUT, int vt_mode) {
  int wid = threadIdx.x >> 5, lane = threadIdx.x & 31;
  int g = lane >> 4, ln = lane & 15;
  int gid = blockIdx.x * 8 + wid;
  int mt = gid >> 4;       // 256 row tiles of 16
  int nt = gid & 15;       // 16 col groups of 64
  int row0 = mt * 16, n0 = nt * 64;

  v8f acc[4] = {};
  const float* xrow = X + (size_t)(row0 + ln) * DD;
  for (int k0 = 0; k0 < DD; k0 += 32) {
    v16bf a;
    ld8f(a, 0, xrow + k0 + g * 8);
    ld8f(a, 8, xrow + k0 + g * 8 + 16);
#pragma unroll
    for (int c = 0; c < 4; ++c) {
      const __bf16* wrow = WT + (size_t)(n0 + c * 16 + ln) * DD + k0 + g * 8;
      v16bf bfr;
      ld8bf(bfr, 0, wrow);
      ld8bf(bfr, 8, wrow + 16);
      acc[c] = WMMA_BF16(a, bfr, acc[c]);
    }
  }

  if (vt_mode) {
    int b  = row0 >> 11;
    int l0 = (row0 & 2047) + g * 8;  // 8 consecutive tokens per lane-half
#pragma unroll
    for (int c = 0; c < 4; ++c) {
      int n = n0 + c * 16 + ln;
      int h = n >> 6, d = n & 63;
      union { uint4 u; __bf16 hh[8]; } t;
#pragma unroll
      for (int r = 0; r < 8; ++r) t.hh[r] = f2bf(acc[c][r]);
      *(uint4*)&OUT[(((size_t)(b * HH + h) * DEPTH + d) * LL) + l0] = t.u;
    }
  } else {
#pragma unroll
    for (int c = 0; c < 4; ++c) {
#pragma unroll
      for (int r = 0; r < 8; ++r) {
        int m = row0 + g * 8 + r;
        int n = n0 + c * 16 + ln;
        int b = m >> 11, l = m & 2047;
        int h = n >> 6, d = n & 63;
        OUT[(((size_t)(b * HH + h) * LL + l) * DEPTH) + d] = f2bf(acc[c][r]);
      }
    }
  }
}

// ---------------------------------------------------------------------------
// Kernel 3: fused flash attention.  One wave owns 16 queries of one (b,h).
// Loops over keys 32 at a time: S = Q K^T (4 wmma), scale + bias*(-1e9),
// online softmax (cross-lane reductions), P through LDS (C->A layout),
// O += P V (4 wmma).  Accumulator is 16x64 fp32 in registers.
// ---------------------------------------------------------------------------
__global__ void __launch_bounds__(256) attn_fused(
    const __bf16* __restrict__ Q, const __bf16* __restrict__ K,
    const __bf16* __restrict__ Vt, const float* __restrict__ bias,
    __bf16* __restrict__ AO) {
  __shared__ __align__(16) __bf16 pbuf[8][16][32];
  int wid = threadIdx.x >> 5, lane = threadIdx.x & 31;
  int g = lane >> 4, ln = lane & 15;
  int gid = blockIdx.x * 8 + wid;
  int bh = gid >> 7;   // 128 query tiles per (b,h)
  int qt = gid & 127;
  int b = bh >> 4, h = bh & 15;
  int q0 = qt * 16;

  const __bf16* Qb = Q  + (size_t)bh * LL * DEPTH;
  const __bf16* Kb = K  + (size_t)bh * LL * DEPTH;
  const __bf16* Vb = Vt + (size_t)bh * DEPTH * LL;
  const float*  Bb = bias + (size_t)b * LL * LL;

  // Q A-fragments for depth slices [0,32) and [32,64) — kept in registers
  v16bf qf[2];
  {
    const __bf16* qp = Qb + (size_t)(q0 + ln) * DEPTH;
#pragma unroll
    for (int s = 0; s < 2; ++s) {
      ld8bf(qf[s], 0, qp + s * 32 + g * 8);
      ld8bf(qf[s], 8, qp + s * 32 + g * 8 + 16);
    }
  }

  v8f o[4] = {};
  float rm[8], rs[8];
#pragma unroll
  for (int r = 0; r < 8; ++r) { rm[r] = -3.0e38f; rs[r] = 0.f; }
  const float scale = 0.125f;  // 1/sqrt(64)

  for (int kb = 0; kb < LL; kb += 32) {
    v8f s0 = {}, s1 = {};
    {  // keys kb..kb+15
      const __bf16* kp = Kb + (size_t)(kb + ln) * DEPTH;
      v16bf kf;
      ld8bf(kf, 0, kp + g * 8);       ld8bf(kf, 8, kp + g * 8 + 16);
      s0 = WMMA_BF16(qf[0], kf, s0);
      ld8bf(kf, 0, kp + 32 + g * 8);  ld8bf(kf, 8, kp + 32 + g * 8 + 16);
      s0 = WMMA_BF16(qf[1], kf, s0);
    }
    {  // keys kb+16..kb+31
      const __bf16* kp = Kb + (size_t)(kb + 16 + ln) * DEPTH;
      v16bf kf;
      ld8bf(kf, 0, kp + g * 8);       ld8bf(kf, 8, kp + g * 8 + 16);
      s1 = WMMA_BF16(qf[0], kf, s1);
      ld8bf(kf, 0, kp + 32 + g * 8);  ld8bf(kf, 8, kp + 32 + g * 8 + 16);
      s1 = WMMA_BF16(qf[1], kf, s1);
    }

    float p0[8], p1[8];
#pragma unroll
    for (int r = 0; r < 8; ++r) {
      int q = q0 + g * 8 + r;
      float b0 = Bb[(size_t)q * LL + kb + ln];
      float b1 = Bb[(size_t)q * LL + kb + 16 + ln];
      float v0 = s0[r] * scale + b0 * NEG_BIG;
      float v1 = s1[r] * scale + b1 * NEG_BIG;
      float tmax = redmax16(fmaxf(v0, v1));
      float nm    = fmaxf(rm[r], tmax);
      float alpha = __expf(rm[r] - nm);
      rm[r] = nm;
      p0[r] = __expf(v0 - nm);
      p1[r] = __expf(v1 - nm);
      float tsum = redsum16(p0[r] + p1[r]);
      rs[r] = rs[r] * alpha + tsum;
#pragma unroll
      for (int c = 0; c < 4; ++c) o[c][r] *= alpha;
    }

    // C-layout P -> LDS (row-major 16x32), then reload as A-fragment
#pragma unroll
    for (int r = 0; r < 8; ++r) {
      pbuf[wid][g * 8 + r][ln]      = f2bf(p0[r]);
      pbuf[wid][g * 8 + r][ln + 16] = f2bf(p1[r]);
    }
    asm volatile("s_wait_dscnt 0" ::: "memory");  // same-wave LDS RAW
    v16bf pf;
    {
      const __bf16* pp = &pbuf[wid][ln][0];
      ld8bf(pf, 0, pp + g * 8);
      ld8bf(pf, 8, pp + g * 8 + 16);
    }

#pragma unroll
    for (int c = 0; c < 4; ++c) {  // output depth columns c*16..c*16+15
      const __bf16* vp = Vb + (size_t)(c * 16 + ln) * LL + kb;
      v16bf vf;
      ld8bf(vf, 0, vp + g * 8);
      ld8bf(vf, 8, vp + g * 8 + 16);
      o[c] = WMMA_BF16(pf, vf, o[c]);
    }
  }

  // normalize and store attn output as bf16 [B*L][D] (heads concatenated)
#pragma unroll
  for (int r = 0; r < 8; ++r) {
    float inv = rs[r] > 0.f ? 1.0f / rs[r] : 0.f;
    int l = q0 + g * 8 + r;
#pragma unroll
    for (int c = 0; c < 4; ++c) {
      int ch = h * DEPTH + c * 16 + ln;
      AO[((size_t)(b * LL + l)) * DD + ch] = f2bf(o[c][r] * inv);
    }
  }
}

// ---------------------------------------------------------------------------
// Kernel 4: output projection  out = fp32( AO[4096x1024] @ Wo )  via Wo^T.
// ---------------------------------------------------------------------------
__global__ void __launch_bounds__(256) out_gemm(
    const __bf16* __restrict__ A, const __bf16* __restrict__ WT,
    float* __restrict__ OUT) {
  int wid = threadIdx.x >> 5, lane = threadIdx.x & 31;
  int g = lane >> 4, ln = lane & 15;
  int gid = blockIdx.x * 8 + wid;
  int mt = gid >> 4;
  int nt = gid & 15;
  int row0 = mt * 16, n0 = nt * 64;

  v8f acc[4] = {};
  const __bf16* arow = A + (size_t)(row0 + ln) * DD;
  for (int k0 = 0; k0 < DD; k0 += 32) {
    v16bf a;
    ld8bf(a, 0, arow + k0 + g * 8);
    ld8bf(a, 8, arow + k0 + g * 8 + 16);
#pragma unroll
    for (int c = 0; c < 4; ++c) {
      const __bf16* wrow = WT + (size_t)(n0 + c * 16 + ln) * DD + k0 + g * 8;
      v16bf bfr;
      ld8bf(bfr, 0, wrow);
      ld8bf(bfr, 8, wrow + 16);
      acc[c] = WMMA_BF16(a, bfr, acc[c]);
    }
  }
#pragma unroll
  for (int c = 0; c < 4; ++c) {
#pragma unroll
    for (int r = 0; r < 8; ++r) {
      int m = row0 + g * 8 + r;
      int n = n0 + c * 16 + ln;
      OUT[(size_t)m * DD + n] = acc[c][r];
    }
  }
}

// ---------------------------------------------------------------------------
// Host launcher.  Workspace layout (bf16):
//   [0, 8MB)   Wq^T, Wk^T, Wv^T, Wo^T  (4 * 1024*1024 * 2B)
//   [8,16MB)   Q   [B,H,L,64]
//   [16,24MB)  K   [B,H,L,64]
//   [24,32MB)  V^T [B,H,64,L]
//   [32,40MB)  attn output [B*L, D]
// ---------------------------------------------------------------------------
extern "C" void kernel_launch(void* const* d_in, const int* in_sizes, int n_in,
                              void* d_out, int out_size, void* d_ws, size_t ws_size,
                              hipStream_t stream) {
  const float* x    = (const float*)d_in[0];
  const float* y    = (const float*)d_in[1];
  const float* bias = (const float*)d_in[2];
  const float* Wq   = (const float*)d_in[3];
  const float* Wk   = (const float*)d_in[4];
  const float* Wv   = (const float*)d_in[5];
  const float* Wo   = (const float*)d_in[6];

  char* ws = (char*)d_ws;
  const size_t wsz  = (size_t)DD * DD;              // elements per weight
  const size_t qkvn = (size_t)BB * HH * LL * DEPTH; // elements per Q/K/Vt
  __bf16* WT  = (__bf16*)ws;
  __bf16* WqT = WT;
  __bf16* WkT = WT + wsz;
  __bf16* WvT = WT + 2 * wsz;
  __bf16* WoT = WT + 3 * wsz;
  __bf16* Qb  = (__bf16*)(ws + 4 * wsz * 2);
  __bf16* Kb  = Qb + qkvn;
  __bf16* Vt  = Kb + qkvn;
  __bf16* AO  = Vt + qkvn;

  // 1) weights -> bf16 transposed
  transpose_w<<<dim3(DD / 32, DD / 32, 4), dim3(32, 8), 0, stream>>>(Wq, Wk, Wv, Wo, WT);
  // 2) projections (Q from x; K,V from y; V stored transposed)
  qkv_gemm<<<512, 256, 0, stream>>>(x, WqT, Qb, 0);
  qkv_gemm<<<512, 256, 0, stream>>>(y, WkT, Kb, 0);
  qkv_gemm<<<512, 256, 0, stream>>>(y, WvT, Vt, 1);
  // 3) fused flash attention
  attn_fused<<<512, 256, 0, stream>>>(Qb, Kb, Vt, bias, AO);
  // 4) output projection -> fp32 d_out
  out_gemm<<<512, 256, 0, stream>>>(AO, WoT, (float*)d_out);
}
